// GCNLayer_5403068859071
// MI455X (gfx1250) — matrix operand order repaired and checked by reading, not compiled, not run on hardware
//
#include <hip/hip_runtime.h>
#include <hip/hip_bf16.h>

typedef __attribute__((ext_vector_type(16))) _Float16 v16h;
typedef __attribute__((ext_vector_type(8)))  float    v8f;

#define N_NODES 50000
#define N_EDGES 640000
#define DFEAT   128

// ---------------------------------------------------------------- utilities

__global__ __launch_bounds__(256) void zero_f4_kernel(float4* p, int n4) {
    int i = blockIdx.x * 256 + threadIdx.x;
    if (i < n4) p[i] = make_float4(0.f, 0.f, 0.f, 0.f);
}

// convert both weight matrices (each 128x256 f32, row-major [out,in]) to f16
__global__ __launch_bounds__(256) void cvt_w_kernel(const float* __restrict__ wm,
                                                    const float* __restrict__ wa,
                                                    _Float16* __restrict__ wmh,
                                                    _Float16* __restrict__ wah) {
    int i = blockIdx.x * 256 + threadIdx.x;   // 0 .. 65535
    if (i < 32768)        wmh[i]        = (_Float16)wm[i];
    else if (i < 65536)   wah[i-32768]  = (_Float16)wa[i-32768];
}

__global__ __launch_bounds__(256) void count_kernel(const int* __restrict__ dst,
                                                    float* __restrict__ counts) {
    int e = blockIdx.x * 256 + threadIdx.x;
    if (e < N_EDGES) atomicAdd(&counts[dst[e]], 1.0f);
}

// ------------------------------------------------------- edge message GEMM
// Per wave: one tile of 16 edges x 128 outputs.
// A[m, 0:256] = concat(nfeats[src[e0+m]], efeats[e0+m]) converted to f16.
// B = W_msg^T (K x N), read from LDS-staged f16 copy of W_msg [128,256].
__global__ __launch_bounds__(256) void edge_msg_kernel(
        const float* __restrict__ nfeats, const float* __restrict__ efeats,
        const _Float16* __restrict__ WmH, const float* __restrict__ bm,
        const int* __restrict__ src, const int* __restrict__ dst,
        float* __restrict__ sums) {
    __shared__ _Float16 Wl[128 * 256];           // 64 KB
    { // cooperative stage of W_msg (f16) into LDS
        const uint4* g = (const uint4*)WmH;
        uint4* l = (uint4*)Wl;
        #pragma unroll
        for (int i = 0; i < 16; ++i)
            l[threadIdx.x + 256 * i] = g[threadIdx.x + 256 * i];
    }
    __syncthreads();

    const int lane = threadIdx.x & 31;
    const int wave = threadIdx.x >> 5;
    const int tile = blockIdx.x * 8 + wave;      // 16-edge tile
    if (tile * 16 >= N_EDGES) return;
    const int e0 = tile * 16;
    const int m  = lane & 15;
    const int hi = lane >> 4;

    const int  e    = e0 + m;
    const int  s    = src[e];
    const float* arow = nfeats + (size_t)s * DFEAT;   // K in [0,128)
    const float* erow = efeats + (size_t)e * DFEAT;   // K-128 in [0,128)

    v8f acc[8];
    #pragma unroll
    for (int nt = 0; nt < 8; ++nt) acc[nt] = (v8f)(0.0f);

    #pragma unroll
    for (int kt = 0; kt < 8; ++kt) {
        const int kbase = kt * 32;
        // A fragment: elems 0..7 -> K = kbase+16*hi+j ; elems 8..15 -> K+8
        const int k0 = kbase + 16 * hi;
        const int k1 = k0 + 8;
        const float* p0 = (k0 < 128) ? (arow + k0) : (erow + (k0 - 128));
        const float* p1 = (k1 < 128) ? (arow + k1) : (erow + (k1 - 128));
        v16h a;
        #pragma unroll
        for (int j = 0; j < 8; ++j) a[j]     = (_Float16)p0[j];
        #pragma unroll
        for (int j = 0; j < 8; ++j) a[8 + j] = (_Float16)p1[j];

        #pragma unroll
        for (int nt = 0; nt < 8; ++nt) {
            // B fragment: lane column n=m, elem e -> K = kbase + 16*hi + e
            const v16h b = *(const v16h*)(Wl + ((nt * 16 + m) * 256 + kbase + 16 * hi));
            acc[nt] = __builtin_amdgcn_wmma_f32_16x16x32_f16(
                          false, a, false, b, (short)0, acc[nt], false, false);
        }
    }

    // scatter: C VGPR r, this lane -> row r+8*hi, col nt*16 + m
    #pragma unroll
    for (int nt = 0; nt < 8; ++nt) {
        const int col = nt * 16 + m;
        const float bias = bm[col];
        #pragma unroll
        for (int r = 0; r < 8; ++r) {
            const int row = r + 8 * hi;
            const int de  = dst[e0 + row];
            atomicAdd(&sums[(size_t)de * DFEAT + col], acc[nt][r] + bias);
        }
    }
}

// ------------------------------------------------------- node apply GEMM
// A[m, 0:256] = concat(nfeats[n0+m], sums[n0+m] / max(counts,1)) as f16.
__global__ __launch_bounds__(256) void node_apply_kernel(
        const float* __restrict__ nfeats, const float* __restrict__ sums,
        const float* __restrict__ counts, const _Float16* __restrict__ WaH,
        const float* __restrict__ ba, float* __restrict__ out) {
    __shared__ _Float16 Wl[128 * 256];
    {
        const uint4* g = (const uint4*)WaH;
        uint4* l = (uint4*)Wl;
        #pragma unroll
        for (int i = 0; i < 16; ++i)
            l[threadIdx.x + 256 * i] = g[threadIdx.x + 256 * i];
    }
    __syncthreads();

    const int lane = threadIdx.x & 31;
    const int wave = threadIdx.x >> 5;
    const int tile = blockIdx.x * 8 + wave;
    if (tile * 16 >= N_NODES) return;
    const int n0 = tile * 16;
    const int m  = lane & 15;
    const int hi = lane >> 4;

    const int node = n0 + m;
    const float inv = 1.0f / fmaxf(counts[node], 1.0f);
    const float* arow = nfeats + (size_t)node * DFEAT;
    const float* srow = sums   + (size_t)node * DFEAT;

    v8f acc[8];
    #pragma unroll
    for (int nt = 0; nt < 8; ++nt) acc[nt] = (v8f)(0.0f);

    #pragma unroll
    for (int kt = 0; kt < 8; ++kt) {
        const int kbase = kt * 32;
        const int k0 = kbase + 16 * hi;
        const int k1 = k0 + 8;
        v16h a;
        if (k1 < 128) {            // both chunks from nfeats
            #pragma unroll
            for (int j = 0; j < 8; ++j) a[j]     = (_Float16)arow[k0 + j];
            #pragma unroll
            for (int j = 0; j < 8; ++j) a[8 + j] = (_Float16)arow[k1 + j];
        } else if (k0 >= 128) {    // both chunks from mean-neighbor
            #pragma unroll
            for (int j = 0; j < 8; ++j) a[j]     = (_Float16)(srow[k0 - 128 + j] * inv);
            #pragma unroll
            for (int j = 0; j < 8; ++j) a[8 + j] = (_Float16)(srow[k1 - 128 + j] * inv);
        } else {                   // straddle (k0 in nfeats, k1 in sums)
            #pragma unroll
            for (int j = 0; j < 8; ++j) a[j]     = (_Float16)arow[k0 + j];
            #pragma unroll
            for (int j = 0; j < 8; ++j) a[8 + j] = (_Float16)(srow[k1 - 128 + j] * inv);
        }
        #pragma unroll
        for (int nt = 0; nt < 8; ++nt) {
            const v16h b = *(const v16h*)(Wl + ((nt * 16 + m) * 256 + kbase + 16 * hi));
            acc[nt] = __builtin_amdgcn_wmma_f32_16x16x32_f16(
                          false, a, false, b, (short)0, acc[nt], false, false);
        }
    }

    #pragma unroll
    for (int nt = 0; nt < 8; ++nt) {
        const int col = nt * 16 + m;
        const float bias = ba[col];
        #pragma unroll
        for (int r = 0; r < 8; ++r) {
            const int row = r + 8 * hi;
            out[(size_t)(n0 + row) * DFEAT + col] = fmaxf(acc[nt][r] + bias, 0.0f);
        }
    }
}

// ---------------------------------------------------------------- launcher

extern "C" void kernel_launch(void* const* d_in, const int* in_sizes, int n_in,
                              void* d_out, int out_size, void* d_ws, size_t ws_size,
                              hipStream_t stream) {
    const float* nfeats = (const float*)d_in[0];
    const float* efeats = (const float*)d_in[1];
    const float* Wmsg   = (const float*)d_in[2];
    const float* bmsg   = (const float*)d_in[3];
    const float* Wapp   = (const float*)d_in[4];
    const float* bapp   = (const float*)d_in[5];
    const int*   src    = (const int*)d_in[6];
    const int*   dst    = (const int*)d_in[7];
    float* out = (float*)d_out;

    // workspace layout
    char* ws = (char*)d_ws;
    float* sums   = (float*)ws;                                  // N*D f32
    float* counts = sums + (size_t)N_NODES * DFEAT;              // N f32
    _Float16* WmH = (_Float16*)(counts + N_NODES);               // 128*256 f16
    _Float16* WaH = WmH + 128 * 256;
    (void)ws_size; (void)in_sizes; (void)n_in; (void)out_size;

    // 1) zero sums + counts
    const int nz4 = (N_NODES * DFEAT + N_NODES) / 4;
    zero_f4_kernel<<<(nz4 + 255) / 256, 256, 0, stream>>>((float4*)sums, nz4);

    // 2) weights -> f16
    cvt_w_kernel<<<65536 / 256, 256, 0, stream>>>(Wmsg, Wapp, WmH, WaH);

    // 3) degree counts
    count_kernel<<<(N_EDGES + 255) / 256, 256, 0, stream>>>(dst, counts);

    // 4) edge message GEMM + scatter-add (40000 16-edge tiles, 8 waves/block)
    edge_msg_kernel<<<(N_EDGES / 16) / 8, 256, 0, stream>>>(
        nfeats, efeats, WmH, bmsg, src, dst, sums);

    // 5) node apply GEMM + mean + ReLU (3125 16-node tiles)
    node_apply_kernel<<<((N_NODES / 16) + 7) / 8, 256, 0, stream>>>(
        nfeats, sums, counts, WaH, bapp, out);
}
